// MoEBlock_40905268527231
// MI455X (gfx1250) — matrix operand order repaired and checked by reading, not compiled, not run on hardware
//
#include <hip/hip_runtime.h>

// ---------------- problem constants ----------------
constexpr int Bb   = 4;
constexpr int Ss   = 2048;
constexpr int Dd   = 1024;
constexpr int Hh   = 512;
constexpr int Ee   = 8;
constexpr int Ff   = 2816;
constexpr int Rr   = 16;
constexpr int HID  = 128;
constexpr int NTOK = Bb * Ss;          // 8192 tokens

constexpr int KTC_D   = Dd / 32;       // 32 k-tiles over D
constexpr int KTC_F   = Ff / 32;       // 88 k-tiles over F
constexpr int KTC_HID = HID / 32;      // 4

// swizzled bf16 weight-cache sizes (elements); one 32x16 tile = 512 elems
constexpr size_t SZ_W1X = (size_t)(HID / 16) * KTC_D * 512;   // 131072
constexpr size_t SZ_W2  = (size_t)(Dd / 16) * KTC_HID * 512;  // 131072
constexpr size_t SZ_WG  = (size_t)(Ff / 16) * KTC_D * 512;    // 2883584
constexpr size_t SZ_WD  = (size_t)(Dd / 16) * KTC_F * 512;    // 2883584
constexpr size_t SZ_AG  = (size_t)1 * KTC_D * 512;            // 16384
constexpr size_t SZ_AD  = (size_t)1 * KTC_F * 512;            // 45056
constexpr size_t SZ_BG  = (size_t)(Ff / 16) * 1 * 512;        // 90112
constexpr size_t SZ_BD  = (size_t)(Dd / 16) * 1 * 512;        // 32768

constexpr size_t OFF_WG = 0;
constexpr size_t OFF_WU = OFF_WG + SZ_WG;
constexpr size_t OFF_WD = OFF_WU + SZ_WG;
constexpr size_t OFF_AG = OFF_WD + SZ_WD;
constexpr size_t OFF_AU = OFF_AG + SZ_AG;
constexpr size_t OFF_AD = OFF_AU + SZ_AG;
constexpr size_t OFF_BG = OFF_AD + SZ_AD;
constexpr size_t OFF_BU = OFF_BG + SZ_BG;
constexpr size_t OFF_BD = OFF_BU + SZ_BG;
constexpr size_t ESTR   = OFF_BD + SZ_BD;                      // 8941568 / expert
constexpr size_t EXP0   = SZ_W1X + SZ_W2;
constexpr size_t TOTAL_BF16 = EXP0 + (size_t)Ee * ESTR;        // ~71.8M elems

// ---------------- bf16 helpers ----------------
typedef __bf16 bf16_t;
typedef bf16_t bf16x16 __attribute__((ext_vector_type(16)));
typedef float  f32x8   __attribute__((ext_vector_type(8)));

static __device__ __forceinline__ bf16_t f2bf(float f) {
  unsigned u = __float_as_uint(f);
  unsigned r = u + 0x7FFFu + ((u >> 16) & 1u);       // RNE
  unsigned short h = (unsigned short)(r >> 16);
  return __builtin_bit_cast(bf16_t, h);
}

static __device__ __forceinline__ f32x8 wmma_bf16(bf16x16 a, bf16x16 b, f32x8 c) {
  return __builtin_amdgcn_wmma_f32_16x16x32_bf16(
      false, a, false, b, (short)0, c, false, false);
}

// ---- fragment-major swizzle -------------------------------------------------
// A 16x32 bf16 tile (ISA 7.12.2): lane(0..15)=row m, holds K {0..7,16..23};
// lane(16..31)=row m, K+8.  Stored lane-major: tile*512 + lane*16 + j.
// element (m, kl) -> lane = (m&15) + 16*hi, hi = (kl>>3)&1, kk = kl-8*hi,
//                    j = ((kk&16)>>1) | (kk&7)
static __device__ __forceinline__ int a_sw_idx(int m, int k, int ktc) {
  int mt = m >> 4, kt = k >> 5, kl = k & 31;
  int hi = (kl >> 3) & 1;
  int kk = kl - (hi << 3);
  int j  = ((kk & 16) >> 1) | (kk & 7);
  return ((mt * ktc + kt) << 9) + ((((m & 15)) + (hi << 4)) << 4) + j;
}

// fast A fragment: one 32-byte LDS read per lane (2x ds_load_b128)
static __device__ __forceinline__ bf16x16 load_a_sw(const bf16_t* mat, int mt, int kt, int ktc) {
  int lane = threadIdx.x & 31;
  return ((const bf16x16*)mat)[(size_t)(mt * ktc + kt) * 32 + lane];
}

// fast B fragment: one 32-byte coalesced global read per lane (2x global_load_b128)
static __device__ __forceinline__ bf16x16 load_b_sw(const bf16_t* mat, int nt, int kt, int ktc) {
  int lane = threadIdx.x & 31;
  return ((const bf16x16*)mat)[(size_t)(nt * ktc + kt) * 32 + lane];
}

// slow-path B fragment from row-major fp32 (on-the-fly convert)
static __device__ __forceinline__ bf16x16 load_b_g(const float* p, int ldb, int kmax) {
  int lane = threadIdx.x & 31;
  int n = lane & 15, hi = lane >> 4;
  bf16x16 r;
#pragma unroll
  for (int j = 0; j < 16; ++j) {
    int k = hi * 8 + (j & 7) + ((j >> 3) << 4);
    float f = (k < kmax) ? p[(size_t)k * ldb + n] : 0.0f;
    r[j] = f2bf(f);
  }
  return r;
}

// stage a row-major fp32 [mtc*16, ktc*32] block into A-swizzled bf16 LDS.
// Each chunk (= one lane's 16 elems) is two 32B runs in the source row.
static __device__ __forceinline__ void stage_a(const float* __restrict__ src, int ld,
                                               bf16_t* dst, int mtc, int ktc) {
  int chunks = mtc * ktc * 32;
  for (int cid = threadIdx.x; cid < chunks; cid += 256) {
    int tile = cid >> 5, lane = cid & 31;
    int kt = tile % ktc, mt = tile / ktc;
    int m = mt * 16 + (lane & 15), hi = lane >> 4;
    int kb = kt * 32 + hi * 8;
    const float4* p0 = (const float4*)(src + (size_t)m * ld + kb);
    const float4* p1 = (const float4*)(src + (size_t)m * ld + kb + 16);
    float4 a = p0[0], b = p0[1], c = p1[0], d = p1[1];
    bf16x16 v;
    v[0] = f2bf(a.x); v[1] = f2bf(a.y); v[2]  = f2bf(a.z); v[3]  = f2bf(a.w);
    v[4] = f2bf(b.x); v[5] = f2bf(b.y); v[6]  = f2bf(b.z); v[7]  = f2bf(b.w);
    v[8] = f2bf(c.x); v[9] = f2bf(c.y); v[10] = f2bf(c.z); v[11] = f2bf(c.w);
    v[12] = f2bf(d.x); v[13] = f2bf(d.y); v[14] = f2bf(d.z); v[15] = f2bf(d.w);
    ((bf16x16*)dst)[tile * 32 + lane] = v;
  }
}

// ============ weight pre-convert: fp32 [K,N] -> swizzled bf16 B tiles ========
__global__ __launch_bounds__(256) void convert_b_kernel(
    const float* __restrict__ src, bf16_t* __restrict__ dst,
    int ktc, int ntc, int ldsrc, int Kvalid, size_t sstride, size_t dstride) {
  int e = blockIdx.y;
  src += (size_t)e * sstride;
  dst += (size_t)e * dstride;
  int cid = blockIdx.x * 256 + threadIdx.x;
  int total = ntc * ktc * 32;
  if (cid >= total) return;
  int tile = cid >> 5, lane = cid & 31;
  int kt = tile % ktc, nt = tile / ktc;
  int n = lane & 15, hi = lane >> 4;
  int col = nt * 16 + n;
  bf16x16 v;
#pragma unroll
  for (int j = 0; j < 16; ++j) {
    int k = kt * 32 + hi * 8 + (j & 7) + ((j >> 3) << 4);
    float f = (k < Kvalid) ? src[(size_t)k * ldsrc + col] : 0.0f;
    v[j] = f2bf(f);
  }
  ((bf16x16*)dst)[(size_t)tile * 32 + lane] = v;
}

// ================= Kernel 0: tiny precompute =================
__global__ __launch_bounds__(128) void moe_prep_kernel(
    const float* __restrict__ hist, const float* __restrict__ persona,
    const float* __restrict__ W1h, const float* __restrict__ W1p,
    const float* __restrict__ ln_g, const float* __restrict__ ln_b,
    const float* __restrict__ Wg,
    float* __restrict__ hh, float* __restrict__ ph, float* __restrict__ cgcb) {
  int blk = blockIdx.x, k = threadIdx.x;
  if (blk < Bb) {
    float s = 0.f;
    for (int i = 0; i < Hh; ++i) s += hist[blk * Hh + i] * W1h[i * HID + k];
    hh[blk * HID + k] = s;
  } else if (blk < Bb + Ee) {
    int e = blk - Bb;
    float s = 0.f;
    for (int i = 0; i < Dd; ++i) s += persona[(size_t)e * Dd + i] * W1p[i * HID + k];
    ph[e * HID + k] = s;
  } else {
    __shared__ float r1[128], r2[128];
    float a = 0.f, b = 0.f;
    for (int d = k; d < Dd; d += 128) {
      float w = Wg[d];
      a += ln_g[d] * w;
      b += ln_b[d] * w;
    }
    r1[k] = a; r2[k] = b;
    __syncthreads();
    for (int off = 64; off > 0; off >>= 1) {
      if (k < off) { r1[k] += r1[k + off]; r2[k] += r2[k + off]; }
      __syncthreads();
    }
    if (k == 0) { cgcb[0] = r1[0]; cgcb[1] = r2[0]; }
  }
}

// ================= Kernel 1: gating (16-token tile, 8 waves) =================
template <bool PRE>
__global__ __launch_bounds__(256) void moe_gating_kernel(
    const float* __restrict__ x,
    const float* __restrict__ W1x, const float* __restrict__ b1,
    const float* __restrict__ W2, const float* __restrict__ b2,
    const float* __restrict__ ln_g, const float* __restrict__ Wg,
    const float* __restrict__ bg,
    const float* __restrict__ hh, const float* __restrict__ ph,
    const float* __restrict__ cgcb, float* __restrict__ gatew,
    const bf16_t* __restrict__ W1xS, const bf16_t* __restrict__ W2S) {
  __shared__ __align__(32) bf16_t xa[16 * Dd];     // A-swizzled x tile (32 KB)
  __shared__ __align__(32) bf16_t h1s[KTC_HID * 512];  // A-swizzled h1 (4 KB)
  __shared__ float xhs[16 * HID];
  __shared__ float red1[16], red2[16], red3[16];
  __shared__ float logits[16 * Ee];

  const int tid = threadIdx.x, lane = tid & 31, wv = tid >> 5;
  const int n = lane & 15, hi = lane >> 4;
  const int tile0 = blockIdx.x * 16;

  stage_a(x + (size_t)tile0 * Dd, Dd, xa, 1, KTC_D);
  __syncthreads();

  // xh = x@W1x  (wave wv -> 16 cols)
  {
    f32x8 acc = {};
    for (int kt = 0; kt < KTC_D; ++kt) {
      bf16x16 a = load_a_sw(xa, 0, kt, KTC_D);
      bf16x16 b = PRE ? load_b_sw(W1xS, wv, kt, KTC_D)
                      : load_b_g(&W1x[(size_t)kt * 32 * HID + wv * 16], HID, 32);
      acc = wmma_bf16(a, b, acc);
    }
#pragma unroll
    for (int r = 0; r < 8; ++r) xhs[(r + 8 * hi) * HID + wv * 16 + n] = acc[r];
  }
  __syncthreads();

  const float Cg = cgcb[0], Cb = cgcb[1], bg0 = bg[0];

  for (int e = 0; e < Ee; ++e) {
    if (tid < 16) { red1[tid] = 0.f; red2[tid] = 0.f; red3[tid] = 0.f; }
    // h1 = relu(xh + hh[b] + ph[e] + b1), written A-swizzled
    for (int i = tid; i < 16 * HID; i += 256) {
      int r = i >> 7, k = i & 127;
      int t = tile0 + r;
      int bi = t >> 11;  // S = 2048
      float v = xhs[i] + hh[bi * HID + k] + ph[e * HID + k] + b1[k];
      h1s[a_sw_idx(r, k, KTC_HID)] = f2bf(v > 0.f ? v : 0.f);
    }
    __syncthreads();

    float s1[8], s2[8], s3[8];
#pragma unroll
    for (int r = 0; r < 8; ++r) { s1[r] = 0.f; s2[r] = 0.f; s3[r] = 0.f; }
    for (int nt = 0; nt < 8; ++nt) {
      int col0 = wv * 128 + nt * 16;
      f32x8 acc = {};
      for (int kt = 0; kt < KTC_HID; ++kt) {
        bf16x16 a = load_a_sw(h1s, 0, kt, KTC_HID);
        bf16x16 b = PRE ? load_b_sw(W2S, col0 >> 4, kt, KTC_HID)
                        : load_b_g(&W2[(size_t)kt * 32 * Dd + col0], Dd, 32);
        acc = wmma_bf16(a, b, acc);
      }
      int col = col0 + n;
      float b2c = b2[col];
      float gwc = ln_g[col] * Wg[col];
#pragma unroll
      for (int r = 0; r < 8; ++r) {
        float f = acc[r] + b2c;
        f = f > 0.f ? f : 0.f;
        s1[r] += f; s2[r] += f * f; s3[r] += f * gwc;
      }
    }
#pragma unroll
    for (int r = 0; r < 8; ++r) {
      int m = r + 8 * hi;
      atomicAdd(&red1[m], s1[r]);
      atomicAdd(&red2[m], s2[r]);
      atomicAdd(&red3[m], s3[r]);
    }
    __syncthreads();
    if (tid < 16) {
      float mu = red1[tid] * (1.0f / Dd);
      float var = red2[tid] * (1.0f / Dd) - mu * mu;
      float rs = rsqrtf(var + 1e-5f);
      logits[tid * Ee + e] = rs * (red3[tid] - mu * Cg) + Cb + bg0;
    }
    __syncthreads();
  }

  if (tid < 16) {
    float l[Ee], mx = -1e30f;
#pragma unroll
    for (int e = 0; e < Ee; ++e) { l[e] = logits[tid * Ee + e]; mx = fmaxf(mx, l[e]); }
    float p[Ee], s = 0.f;
#pragma unroll
    for (int e = 0; e < Ee; ++e) { p[e] = __expf(l[e] - mx); s += p[e]; }
    int i1 = 0;
#pragma unroll
    for (int e = 1; e < Ee; ++e) if (p[e] > p[i1]) i1 = e;
    int i2 = (i1 == 0) ? 1 : 0;
#pragma unroll
    for (int e = 0; e < Ee; ++e) { if (e != i1 && p[e] > p[i2]) i2 = e; }
    int t = tile0 + tid;
    float inv = 1.0f / s;
#pragma unroll
    for (int e = 0; e < Ee; ++e)
      gatew[(size_t)t * Ee + e] = (e == i1 || e == i2) ? p[e] * inv : 0.f;
  }
}

// ================= Kernel 2: fused expert FFN (32-token tile x expert) =================
template <bool PRE>
__global__ __launch_bounds__(256) void moe_expert_kernel(
    const float* __restrict__ x, const float* __restrict__ gatew,
    const float* __restrict__ Wgate, const float* __restrict__ Wup,
    const float* __restrict__ Wdown,
    const float* __restrict__ Ag, const float* __restrict__ Bg,
    const float* __restrict__ Au, const float* __restrict__ Bu,
    const float* __restrict__ Ad, const float* __restrict__ Bd,
    const bf16_t* __restrict__ wexp,   // swizzled bf16 expert block base (EXP0 offset applied)
    float* __restrict__ out) {
  __shared__ __align__(32) bf16_t xa[32 * Dd];          // A-swizzled x tile (64 KB)
  __shared__ __align__(32) bf16_t hbuf[2 * KTC_F * 512]; // A-swizzled h (176 KB)
  __shared__ __align__(32) bf16_t tg[2 * 512];          // x@Ag (K padded to 32)
  __shared__ __align__(32) bf16_t tu[2 * 512];
  __shared__ __align__(32) bf16_t td[2 * 512];
  __shared__ float wrow[32];
  __shared__ int anyflag;

  const int e = blockIdx.y;
  const int tile0 = blockIdx.x * 32;
  const int tid = threadIdx.x, lane = tid & 31, wv = tid >> 5;
  const int n = lane & 15, hi = lane >> 4;
  const bf16_t BZ = __builtin_bit_cast(bf16_t, (unsigned short)0);

  if (tid == 0) anyflag = 0;
  __syncthreads();
  if (tid < 32) {
    float w = gatew[(size_t)(tile0 + tid) * Ee + e];
    wrow[tid] = w;
    if (w != 0.f) atomicOr(&anyflag, 1);
  }
  __syncthreads();
  if (!anyflag) return;

  // staging: x tile + zero the LoRA tiles (padding rows must be 0)
  stage_a(x + (size_t)tile0 * Dd, Dd, xa, 2, KTC_D);
  for (int i = tid; i < 2 * 512; i += 256) { tg[i] = BZ; tu[i] = BZ; td[i] = BZ; }
  __syncthreads();

  const bf16_t* WE = wexp + (size_t)e * ESTR;
  const float* WgE = Wgate + (size_t)e * Dd * Ff;
  const float* WuE = Wup + (size_t)e * Dd * Ff;
  const float* WdE = Wdown + (size_t)e * Ff * Dd;
  const float* AgE = Ag + (size_t)e * Dd * Rr;
  const float* AuE = Au + (size_t)e * Dd * Rr;
  const float* AdE = Ad + (size_t)e * Ff * Rr;
  const float* BgE = Bg + (size_t)e * Rr * Ff;
  const float* BuE = Bu + (size_t)e * Rr * Ff;
  const float* BdE = Bd + (size_t)e * Rr * Dd;

  // LoRA pre-projection tg = x@Ag, tu = x@Au (waves 0..3)
  if (wv < 4) {
    const bf16_t* AS = (wv >> 1) ? (WE + OFF_AU) : (WE + OFF_AG);
    const float* AF = (wv >> 1) ? AuE : AgE;
    bf16_t* tdst = (wv >> 1) ? tu : tg;
    int half = wv & 1;
    f32x8 acc = {};
    for (int kt = 0; kt < KTC_D; ++kt) {
      bf16x16 a = load_a_sw(xa, half, kt, KTC_D);
      bf16x16 b = PRE ? load_b_sw(AS, 0, kt, KTC_D)
                      : load_b_g(&AF[(size_t)kt * 32 * Rr], Rr, 32);
      acc = wmma_bf16(a, b, acc);
    }
    // store into A-swizzled [32 x 32] tile (cols = R index, kl = n, j = n&7)
#pragma unroll
    for (int r = 0; r < 8; ++r) {
      int idx = half * 512 + ((r + 8 * hi + ((n >> 3) << 4)) << 4) + (n & 7);
      tdst[idx] = f2bf(acc[r]);
    }
  }
  __syncthreads();

  // Stage 1: h = silu(x@Wg + lora) * (x@Wu + lora) -> hbuf (A-swizzled over [32, F])
  constexpr int FPW = Ff / 8;  // 352 cols / wave
  for (int nt = 0; nt < FPW / 16; ++nt) {
    int col0 = wv * FPW + nt * 16;
    int ntg = col0 >> 4;
    f32x8 ag0 = {}, ag1 = {}, au0 = {}, au1 = {};
    for (int kt = 0; kt < KTC_D; ++kt) {
      bf16x16 a0 = load_a_sw(xa, 0, kt, KTC_D);
      bf16x16 a1 = load_a_sw(xa, 1, kt, KTC_D);
      bf16x16 bgf, buf;
      if (PRE) {
        const bf16x16* pg = (const bf16x16*)(WE + OFF_WG) + (size_t)(ntg * KTC_D + kt) * 32;
        const bf16x16* pu = (const bf16x16*)(WE + OFF_WU) + (size_t)(ntg * KTC_D + kt) * 32;
        bgf = pg[lane];
        buf = pu[lane];
        __builtin_prefetch((const void*)(pg + 64), 0, 1);  // next-next k-tile
        __builtin_prefetch((const void*)(pu + 64), 0, 1);
      } else {
        bgf = load_b_g(&WgE[(size_t)kt * 32 * Ff + col0], Ff, 32);
        buf = load_b_g(&WuE[(size_t)kt * 32 * Ff + col0], Ff, 32);
      }
      ag0 = wmma_bf16(a0, bgf, ag0);
      ag1 = wmma_bf16(a1, bgf, ag1);
      au0 = wmma_bf16(a0, buf, au0);
      au1 = wmma_bf16(a1, buf, au1);
    }
    {  // rank-16 LoRA (tiles already zero-padded to K=32)
      bf16x16 a0 = load_a_sw(tg, 0, 0, 1);
      bf16x16 a1 = load_a_sw(tg, 1, 0, 1);
      bf16x16 bb = PRE ? load_b_sw(WE + OFF_BG, ntg, 0, 1) : load_b_g(&BgE[col0], Ff, 16);
      ag0 = wmma_bf16(a0, bb, ag0);
      ag1 = wmma_bf16(a1, bb, ag1);
      a0 = load_a_sw(tu, 0, 0, 1);
      a1 = load_a_sw(tu, 1, 0, 1);
      bb = PRE ? load_b_sw(WE + OFF_BU, ntg, 0, 1) : load_b_g(&BuE[col0], Ff, 16);
      au0 = wmma_bf16(a0, bb, au0);
      au1 = wmma_bf16(a1, bb, au1);
    }
    // write h into A-swizzled hbuf; per-lane column constants
    int c = col0 + n;
    int kt2 = c >> 5, kl = c & 31;
    int hi2 = (kl >> 3) & 1;
    int kk = kl - (hi2 << 3);
    int jj = ((kk & 16) >> 1) | (kk & 7);
    int base0 = (kt2 << 9) + (hi2 << 8) + jj;              // mt=0 tile
    int base1 = ((KTC_F + kt2) << 9) + (hi2 << 8) + jj;    // mt=1 tile
#pragma unroll
    for (int r = 0; r < 8; ++r) {
      int mL = r + 8 * hi;
      float g0 = ag0[r], u0 = au0[r];
      float g1 = ag1[r], u1 = au1[r];
      float h0 = (g0 / (1.f + __expf(-g0))) * u0;
      float h1 = (g1 / (1.f + __expf(-g1))) * u1;
      hbuf[base0 + (mL << 4)] = f2bf(h0);
      hbuf[base1 + (mL << 4)] = f2bf(h1);
    }
  }
  __syncthreads();

  // LoRA down pre-projection td = h@Ad (waves 0..1)
  if (wv < 2) {
    int half = wv;
    f32x8 acc = {};
    for (int kt = 0; kt < KTC_F; ++kt) {
      bf16x16 a = load_a_sw(hbuf, half, kt, KTC_F);
      bf16x16 b = PRE ? load_b_sw(WE + OFF_AD, 0, kt, KTC_F)
                      : load_b_g(&AdE[(size_t)kt * 32 * Rr], Rr, 32);
      acc = wmma_bf16(a, b, acc);
    }
#pragma unroll
    for (int r = 0; r < 8; ++r) {
      int idx = half * 512 + ((r + 8 * hi + ((n >> 3) << 4)) << 4) + (n & 7);
      td[idx] = f2bf(acc[r]);
    }
  }
  __syncthreads();

  // Stage 2: o = h@Wd + lora; out += w_e * o
  for (int nt = 0; nt < 8; ++nt) {
    int col0 = wv * 128 + nt * 16;
    int ntg = col0 >> 4;
    f32x8 c0 = {}, c1 = {};
    for (int kt = 0; kt < KTC_F; ++kt) {
      bf16x16 a0 = load_a_sw(hbuf, 0, kt, KTC_F);
      bf16x16 a1 = load_a_sw(hbuf, 1, kt, KTC_F);
      bf16x16 b;
      if (PRE) {
        const bf16x16* pd = (const bf16x16*)(WE + OFF_WD) + (size_t)(ntg * KTC_F + kt) * 32;
        b = pd[lane];
        __builtin_prefetch((const void*)(pd + 64), 0, 1);
      } else {
        b = load_b_g(&WdE[(size_t)kt * 32 * Dd + col0], Dd, 32);
      }
      c0 = wmma_bf16(a0, b, c0);
      c1 = wmma_bf16(a1, b, c1);
    }
    {
      bf16x16 a0 = load_a_sw(td, 0, 0, 1);
      bf16x16 a1 = load_a_sw(td, 1, 0, 1);
      bf16x16 b = PRE ? load_b_sw(WE + OFF_BD, ntg, 0, 1) : load_b_g(&BdE[col0], Dd, 16);
      c0 = wmma_bf16(a0, b, c0);
      c1 = wmma_bf16(a1, b, c1);
    }
#pragma unroll
    for (int r = 0; r < 8; ++r) {
      int m0 = r + 8 * hi, m1 = 16 + m0;
      float v0 = c0[r] * wrow[m0];
      float v1 = c1[r] * wrow[m1];
      if (v0 != 0.f) unsafeAtomicAdd(&out[(size_t)(tile0 + m0) * Dd + col0 + n], v0);
      if (v1 != 0.f) unsafeAtomicAdd(&out[(size_t)(tile0 + m1) * Dd + col0 + n], v1);
    }
  }
}

// ================= host-side launch =================
extern "C" void kernel_launch(void* const* d_in, const int* in_sizes, int n_in,
                              void* d_out, int out_size, void* d_ws, size_t ws_size,
                              hipStream_t stream) {
  const float* x       = (const float*)d_in[0];
  const float* hist    = (const float*)d_in[1];
  const float* persona = (const float*)d_in[2];
  const float* W1x     = (const float*)d_in[3];
  const float* W1h     = (const float*)d_in[4];
  const float* W1p     = (const float*)d_in[5];
  const float* b1      = (const float*)d_in[6];
  const float* W2      = (const float*)d_in[7];
  const float* b2      = (const float*)d_in[8];
  const float* ln_g    = (const float*)d_in[9];
  const float* ln_b    = (const float*)d_in[10];
  const float* Wg      = (const float*)d_in[11];
  const float* bg      = (const float*)d_in[12];
  const float* Wgate   = (const float*)d_in[13];
  const float* Wup     = (const float*)d_in[14];
  const float* Wdown   = (const float*)d_in[15];
  const float* Ag      = (const float*)d_in[16];
  const float* Bg      = (const float*)d_in[17];
  const float* Au      = (const float*)d_in[18];
  const float* Bu      = (const float*)d_in[19];
  const float* Ad      = (const float*)d_in[20];
  const float* Bd      = (const float*)d_in[21];
  float* out = (float*)d_out;

  float* wsf   = (float*)d_ws;
  float* hh    = wsf;
  float* ph    = hh + Bb * HID;
  float* cgcb  = ph + Ee * HID;
  float* gatew = cgcb + 2;
  size_t fcount = (size_t)Bb * HID + Ee * HID + 2 + (size_t)NTOK * Ee;
  size_t fbytes = (fcount * sizeof(float) + 31) & ~(size_t)31;
  bf16_t* wsb = (bf16_t*)((char*)d_ws + fbytes);
  bool pre = ws_size >= fbytes + TOTAL_BF16 * sizeof(bf16_t);

  moe_prep_kernel<<<Bb + Ee + 1, 128, 0, stream>>>(hist, persona, W1h, W1p,
                                                   ln_g, ln_b, Wg, hh, ph, cgcb);

  if (pre) {
    auto conv = [&](const float* src, bf16_t* dst, int Kpad, int Kvalid, int N,
                    int ldsrc, size_t sstride, size_t dstride, int ne) {
      int ktc = Kpad / 32, ntc = N / 16;
      int chunks = ntc * ktc * 32;
      convert_b_kernel<<<dim3((chunks + 255) / 256, ne), 256, 0, stream>>>(
          src, dst, ktc, ntc, ldsrc, Kvalid, sstride, dstride);
    };
    bf16_t* EB = wsb + EXP0;
    conv(W1x, wsb, Dd, Dd, HID, HID, 0, 0, 1);
    conv(W2, wsb + SZ_W1X, HID, HID, Dd, Dd, 0, 0, 1);
    conv(Wgate, EB + OFF_WG, Dd, Dd, Ff, Ff, (size_t)Dd * Ff, ESTR, Ee);
    conv(Wup,   EB + OFF_WU, Dd, Dd, Ff, Ff, (size_t)Dd * Ff, ESTR, Ee);
    conv(Wdown, EB + OFF_WD, Ff, Ff, Dd, Dd, (size_t)Ff * Dd, ESTR, Ee);
    conv(Ag, EB + OFF_AG, Dd, Dd, Rr, Rr, (size_t)Dd * Rr, ESTR, Ee);
    conv(Au, EB + OFF_AU, Dd, Dd, Rr, Rr, (size_t)Dd * Rr, ESTR, Ee);
    conv(Ad, EB + OFF_AD, Ff, Ff, Rr, Rr, (size_t)Ff * Rr, ESTR, Ee);
    conv(Bg, EB + OFF_BG, 32, Rr, Ff, Ff, (size_t)Rr * Ff, ESTR, Ee);
    conv(Bu, EB + OFF_BU, 32, Rr, Ff, Ff, (size_t)Rr * Ff, ESTR, Ee);
    conv(Bd, EB + OFF_BD, 32, Rr, Dd, Dd, (size_t)Rr * Dd, ESTR, Ee);
  }

  hipMemsetAsync(d_out, 0, (size_t)out_size * sizeof(float), stream);

  if (pre) {
    moe_gating_kernel<true><<<NTOK / 16, 256, 0, stream>>>(
        x, W1x, b1, W2, b2, ln_g, Wg, bg, hh, ph, cgcb, gatew, wsb, wsb + SZ_W1X);
    moe_expert_kernel<true><<<dim3(NTOK / 32, Ee), 256, 0, stream>>>(
        x, gatew, Wgate, Wup, Wdown, Ag, Bg, Au, Bu, Ad, Bd, wsb + EXP0, out);
  } else {
    moe_gating_kernel<false><<<NTOK / 16, 256, 0, stream>>>(
        x, W1x, b1, W2, b2, ln_g, Wg, bg, hh, ph, cgcb, gatew, wsb, wsb + SZ_W1X);
    moe_expert_kernel<false><<<dim3(NTOK / 32, Ee), 256, 0, stream>>>(
        x, gatew, Wgate, Wup, Wdown, Ag, Bg, Au, Bu, Ad, Bd, wsb + EXP0, out);
  }
}
